// GraphNN_70755291234308
// MI455X (gfx1250) — matrix operand order repaired and checked by reading, not compiled, not run on hardware
//
#include <hip/hip_runtime.h>
#include <hip/hip_bf16.h>

// ---------------------------------------------------------------------------
// 2-layer GCN for MI455X (gfx1250, wave32).
//  * GEMMs: v_wmma_f32_16x16x32_bf16, K fully unrolled (templated), A loaded
//    as b128, W staged into LDS via global_load_async_to_lds_b128 (ASYNCcnt)
//    and converted f32->bf16 with native casts at fragment-read time.
//  * Aggregation: edge-parallel f32 atomics; all operands L2-resident (192MB).
// ---------------------------------------------------------------------------

#define N_NODES 50000

typedef __attribute__((ext_vector_type(16))) __bf16 v16bf;
typedef __attribute__((ext_vector_type(8)))  float  v8f;

// ---------------- degree / normalization ----------------
__global__ void k_deg_init(float* __restrict__ deg, int n) {
  int i = blockIdx.x * blockDim.x + threadIdx.x;
  if (i < n) deg[i] = 1.0f;                          // self-loop contributes 1
}

__global__ void k_deg_count(const int* __restrict__ dst, float* __restrict__ deg, int E) {
  int e = blockIdx.x * blockDim.x + threadIdx.x;
  if (e < E) unsafeAtomicAdd(&deg[dst[e]], 1.0f);
}

__global__ void k_dinv(float* __restrict__ deg, int n) {
  int i = blockIdx.x * blockDim.x + threadIdx.x;
  if (i < n) deg[i] = rsqrtf(deg[i]);                // deg >= 1 always
}

// ---------------- GEMM: C[M,N] = A[M,K] * W[K,N], bf16 WMMA ----------------
// block = 256 threads = 8 waves; wave w computes the 16x16 tile at
// (blockIdx.x*16, blockIdx.y*128 + w*16). The block's 128 W columns are kept
// in LDS as raw f32 in global (row-major) layout, staged asynchronously.
// B-fragment reads are conflict-free: for fixed k, lanes hit 16 consecutive
// dwords.
template <int K, int N>
__global__ void gcn_gemm_bf16(const float* __restrict__ A, const float* __restrict__ W,
                              float* __restrict__ C) {
  extern __shared__ float smemf[];                   // [K][128] f32
  const int n0_block = blockIdx.y * 128;

  // --- async stage: LDS[k][nl] = W[k][n0_block + nl], 16B per thread/iter ---
  {
    unsigned lds_base = (unsigned)(size_t)smemf;     // LDS byte offset of smemf
    for (int idx = threadIdx.x; idx < K * 32; idx += 256) {
      int c = (idx & 31) << 2;                       // column (floats), /16B chunk
      int k = idx >> 5;
      const float* gp = W + (size_t)k * N + n0_block + c;
      unsigned loff = lds_base + (unsigned)((k * 128 + c) * 4);
      asm volatile("global_load_async_to_lds_b128 %0, %1, off"
                   :: "v"(loff), "v"(gp) : "memory");
    }
    asm volatile("s_wait_asynccnt 0x0" ::: "memory");
  }
  __syncthreads();

  const int wave   = threadIdx.x >> 5;
  const int lane   = threadIdx.x & 31;
  const int laneLo = lane & 15;
  const bool hi    = (lane & 16) != 0;
  const int n0     = n0_block + wave * 16;
  const int m      = blockIdx.x * 16 + laneLo;       // M == 50000 = 3125*16
  const int nl     = wave * 16 + laneLo;             // column within LDS block

  const float* Arow = A + (size_t)m * K;

  v8f c = {};
#pragma unroll
  for (int k0 = 0; k0 < K; k0 += 32) {
    v16bf a, b;
    // A fragment (16-bit A 16x32): lanes 0-15 K={0..7,16..23}, lanes 16-31 +8
    const int ka = k0 + (hi ? 8 : 0);
    float4 p0 = *(const float4*)(Arow + ka);
    float4 p1 = *(const float4*)(Arow + ka + 4);
    float4 q0 = *(const float4*)(Arow + ka + 16);
    float4 q1 = *(const float4*)(Arow + ka + 20);
    a[0]  = (__bf16)p0.x; a[1]  = (__bf16)p0.y; a[2]  = (__bf16)p0.z; a[3]  = (__bf16)p0.w;
    a[4]  = (__bf16)p1.x; a[5]  = (__bf16)p1.y; a[6]  = (__bf16)p1.z; a[7]  = (__bf16)p1.w;
    a[8]  = (__bf16)q0.x; a[9]  = (__bf16)q0.y; a[10] = (__bf16)q0.z; a[11] = (__bf16)q0.w;
    a[12] = (__bf16)q1.x; a[13] = (__bf16)q1.y; a[14] = (__bf16)q1.z; a[15] = (__bf16)q1.w;
    // B fragment: same K order as A: lanes 0-15 K={0..7,16..23}, lanes 16-31 +8
    const int kb = k0 + (hi ? 8 : 0);
#pragma unroll
    for (int j = 0; j < 16; ++j)
      b[j] = (__bf16)smemf[(kb + j + (j >= 8 ? 8 : 0)) * 128 + nl];

    c = __builtin_amdgcn_wmma_f32_16x16x32_bf16(false, a, false, b,
                                                (short)0, c, false, false);
  }

  // C layout: VGPR r -> row r (lanes 0-15) / row r+8 (lanes 16-31), col = laneLo
  const int mBase = blockIdx.x * 16 + (hi ? 8 : 0);
#pragma unroll
  for (int r = 0; r < 8; ++r)
    C[(size_t)(mBase + r) * N + n0 + laneLo] = c[r];
}

// ---------------- self-loop + bias: out = h * dinv[i]^2 + bias ----------------
template <int F>
__global__ void k_self_bias(const float* __restrict__ h, const float* __restrict__ dinv,
                            const float* __restrict__ bias, float* __restrict__ out,
                            long long total4) {
  long long t = (long long)blockIdx.x * blockDim.x + threadIdx.x;
  if (t >= total4) return;
  long long idx = t * 4;
  int i = (int)(idx / F);                            // constexpr F -> shifts
  int f = (int)(idx % F);
  float d = dinv[i];
  float s = d * d;
  float4 hv = *(const float4*)(h + idx);
  float4 bv = *(const float4*)(bias + f);
  float4 o;
  o.x = hv.x * s + bv.x; o.y = hv.y * s + bv.y;
  o.z = hv.z * s + bv.z; o.w = hv.w * s + bv.w;
  *(float4*)(out + idx) = o;
}

// ---------------- edge scatter: out[dst] += h[src] * dinv[src]*dinv[dst] ------
template <int F>
__global__ void k_edge_scatter(const int* __restrict__ src, const int* __restrict__ dst,
                               const float* __restrict__ dinv, const float* __restrict__ h,
                               float* __restrict__ out, int E) {
  constexpr int TPE = F >> 2;                        // threads per edge (float4 each)
  long long t = (long long)blockIdx.x * blockDim.x + threadIdx.x;
  int e = (int)(t / TPE);                            // constexpr -> shift
  int c = (int)(t % TPE) * 4;
  if (e >= E) return;
  int s = src[e], d = dst[e];
  float norm = dinv[s] * dinv[d];
  float4 hv = *(const float4*)(h + (size_t)s * F + c);
  float* o = out + (size_t)d * F + c;
  unsafeAtomicAdd(o + 0, hv.x * norm);
  unsafeAtomicAdd(o + 1, hv.y * norm);
  unsafeAtomicAdd(o + 2, hv.z * norm);
  unsafeAtomicAdd(o + 3, hv.w * norm);
}

// ---------------------------------------------------------------------------
extern "C" void kernel_launch(void* const* d_in, const int* in_sizes, int n_in,
                              void* d_out, int out_size, void* d_ws, size_t ws_size,
                              hipStream_t stream) {
  const float* emb = (const float*)d_in[0];
  const float* W1  = (const float*)d_in[1];
  const float* b1  = (const float*)d_in[2];
  const float* W2  = (const float*)d_in[3];
  const float* b2  = (const float*)d_in[4];
  const int*   eix = (const int*)d_in[5];
  const int E = in_sizes[5] / 2;
  const int* src = eix;
  const int* dst = eix + E;

  const int N = N_NODES;
  constexpr int F1 = 256;      // hidden width
  constexpr int F0 = 128;      // in/out width

  // workspace layout (h2 reuses the h_lin region; ~98 MB total)
  char* ws   = (char*)d_ws;
  float* dinv = (float*)(ws);                                     // N floats
  float* hlin = (float*)(ws + 204800);                            // N*256 f32
  float* hagg = (float*)(ws + 204800 + (size_t)N * F1 * 4);       // N*256 f32
  float* h2   = hlin;                                             // N*128 f32 (reuse)
  float* out  = (float*)d_out;                                    // N*128 f32

  // --- normalization ---
  k_deg_init<<<(N + 255) / 256, 256, 0, stream>>>(dinv, N);
  k_deg_count<<<(E + 255) / 256, 256, 0, stream>>>(dst, dinv, E);
  k_dinv<<<(N + 255) / 256, 256, 0, stream>>>(dinv, N);

  // --- layer 1: h = GCNConv(emb; W1, b1) ---
  {
    dim3 g(N / 16, F1 / 128);
    gcn_gemm_bf16<F0, F1><<<g, 256, (size_t)F0 * 128 * 4, stream>>>(emb, W1, hlin);

    long long tot4 = (long long)N * F1 / 4;
    k_self_bias<F1><<<(unsigned)((tot4 + 255) / 256), 256, 0, stream>>>(hlin, dinv, b1, hagg, tot4);

    long long te = (long long)E * (F1 / 4);
    k_edge_scatter<F1><<<(unsigned)((te + 255) / 256), 256, 0, stream>>>(src, dst, dinv, hlin, hagg, E);
  }

  // --- layer 2: out = GCNConv(h; W2, b2) ---
  {
    dim3 g(N / 16, F0 / 128);
    gcn_gemm_bf16<F1, F0><<<g, 256, (size_t)F1 * 128 * 4, stream>>>(hagg, W2, h2);

    long long tot4 = (long long)N * F0 / 4;
    k_self_bias<F0><<<(unsigned)((tot4 + 255) / 256), 256, 0, stream>>>(h2, dinv, b2, out, tot4);

    long long te = (long long)E * (F0 / 4);
    k_edge_scatter<F0><<<(unsigned)((te + 255) / 256), 256, 0, stream>>>(src, dst, dinv, h2, out, E);
  }
}
